// PointNetSetAbstraction_27041114096401
// MI455X (gfx1250) — matrix operand order repaired and checked by reading, not compiled, not run on hardware
//
#include <hip/hip_runtime.h>
#include <hip/hip_bf16.h>

typedef _Float16 v16h __attribute__((ext_vector_type(16)));
typedef _Float16 v8h  __attribute__((ext_vector_type(8)));
typedef float    v8f  __attribute__((ext_vector_type(8)));

#define BB   16
#define NN   4096
#define SS   1024
#define KNN  32
#define CIN  64
#define CF   67          // 3 + CIN
#define C0P  96          // layer-0 K padded to 3*32
#define M_TOT (BB*SS*KNN)  // 524288
#define RAD2 (0.2f*0.2f)
#define BN_EPS 1e-5f

// ---------------------------------------------------------------------------
// 1) Farthest point sampling: one block (1024 thr) per batch, serial over S.
// ---------------------------------------------------------------------------
__global__ __launch_bounds__(1024)
void fps_kernel(const float* __restrict__ x, int* __restrict__ fps_idx) {
  __shared__ float dists[NN];
  __shared__ float wval[32];
  __shared__ int   widx[32];
  __shared__ int   s_far;
  const int b = blockIdx.x;
  const int t = threadIdx.x;
  const int lane = t & 31, wv = t >> 5;
  const float* xb = x + (size_t)b * NN * 3;
  for (int i = t; i < NN; i += 1024) dists[i] = 1e10f;
  if (t == 0) s_far = 0;
  __syncthreads();
  float px[4], py[4], pz[4];
#pragma unroll
  for (int r = 0; r < 4; ++r) {
    int i = t + r * 1024;
    px[r] = xb[i*3+0]; py[r] = xb[i*3+1]; pz[r] = xb[i*3+2];
  }
  for (int s = 0; s < SS; ++s) {
    int far = s_far;
    if (t == 0) fps_idx[b * SS + s] = far;
    float cx = xb[far*3+0], cy = xb[far*3+1], cz = xb[far*3+2];
    float bv = -1.0f; int bi = 0;
#pragma unroll
    for (int r = 0; r < 4; ++r) {
      int i = t + r * 1024;
      float dx = px[r]-cx, dy = py[r]-cy, dz = pz[r]-cz;
      float d  = dx*dx + dy*dy + dz*dz;
      float nd = fminf(dists[i], d);
      dists[i] = nd;
      if (nd > bv) { bv = nd; bi = i; }
    }
#pragma unroll
    for (int off = 16; off > 0; off >>= 1) {
      float ov = __shfl_xor(bv, off, 32);
      int   oi = __shfl_xor(bi, off, 32);
      if (ov > bv || (ov == bv && oi < bi)) { bv = ov; bi = oi; }
    }
    if (lane == 0) { wval[wv] = bv; widx[wv] = bi; }
    __syncthreads();
    if (wv == 0) {
      float v = wval[lane]; int ix = widx[lane];
#pragma unroll
      for (int off = 16; off > 0; off >>= 1) {
        float ov = __shfl_xor(v, off, 32);
        int   oi = __shfl_xor(ix, off, 32);
        if (ov > v || (ov == v && oi < ix)) { v = ov; ix = oi; }
      }
      if (lane == 0) s_far = ix;
    }
    __syncthreads();
  }
}

// ---------------------------------------------------------------------------
// 2) Ball query (first-K by index, wave32 ballot scan) + gather + build
//    X0 f16 [M, 96] (cols 0-2 centered xyz, 3-66 features, 67-95 zero pad).
//    One wave per centroid.
// ---------------------------------------------------------------------------
__global__ __launch_bounds__(256)
void group_gather_kernel(const float* __restrict__ x,
                         const float* __restrict__ xc,
                         const int*   __restrict__ fps_idx,
                         _Float16*    __restrict__ X0) {
  __shared__ int idxbuf[8][KNN];
  const int wave = threadIdx.x >> 5;
  const int lane = threadIdx.x & 31;
  const int wid  = blockIdx.x * 8 + wave;          // 0..B*S-1
  const int b = wid / SS;
  const float* xb = x + (size_t)b * NN * 3;
  const int ci = fps_idx[wid];
  const float cx = xb[ci*3+0], cy = xb[ci*3+1], cz = xb[ci*3+2];
  int cnt = 0;
  for (int base = 0; base < NN && cnt < KNN; base += 32) {
    int i = base + lane;
    float dx = xb[i*3+0]-cx, dy = xb[i*3+1]-cy, dz = xb[i*3+2]-cz;
    float d2 = dx*dx + dy*dy + dz*dz;
    bool in = (d2 <= RAD2);
    unsigned m = (unsigned)__ballot(in);
    int pos = cnt + __popc(m & ((1u << lane) - 1u));
    if (in && pos < KNN) idxbuf[wave][pos] = i;
    cnt += __popc(m);
  }
  if (cnt < KNN) {                      // centroid itself always matches -> cnt>=1
    int first = idxbuf[wave][0];
    if (lane >= cnt && lane < KNN) idxbuf[wave][lane] = first;
  }
  // gather: one point (row) per lane
  const int pi = idxbuf[wave][lane];
  _Float16* dst = X0 + ((size_t)wid * KNN + lane) * C0P;
  _Float16 tmp[C0P];
  tmp[0] = (_Float16)(xb[pi*3+0] - cx);
  tmp[1] = (_Float16)(xb[pi*3+1] - cy);
  tmp[2] = (_Float16)(xb[pi*3+2] - cz);
  const float* fp = xc + ((size_t)b * NN + pi) * CIN;
#pragma unroll
  for (int c = 0; c < CIN; ++c) tmp[3 + c] = (_Float16)fp[c];
#pragma unroll
  for (int c = 3 + CIN; c < C0P; ++c) tmp[c] = (_Float16)0.0f;
#pragma unroll
  for (int v = 0; v < C0P / 8; ++v)
    *(v8h*)(dst + v * 8) = *(const v8h*)(tmp + v * 8);
}

// ---------------------------------------------------------------------------
// WMMA fragment helpers (layouts per CDNA5 ISA 7.12.2)
// ---------------------------------------------------------------------------
__device__ __forceinline__ v16h pack_a(v8h lo, v8h hi) {
  v16h a;
#pragma unroll
  for (int j = 0; j < 8; ++j) { a[j] = lo[j]; a[8 + j] = hi[j]; }
  return a;
}

// ---------------------------------------------------------------------------
// 3) Layer 0: Y0[M,64] = X0[M,96(f16)] * W0^T + b0 ; accumulate BN stats.
//    8 waves/block, 16 rows/wave.
// ---------------------------------------------------------------------------
__global__ __launch_bounds__(256)
void gemm0_kernel(const _Float16* __restrict__ X0,
                  const float* __restrict__ W0,   // [64][67]
                  const float* __restrict__ bias,
                  float* __restrict__ Yout,       // [M][64]
                  float* __restrict__ sums, float* __restrict__ sqs) {
  __shared__ __align__(32) _Float16 Wl[64 * C0P];
  const int t = threadIdx.x;
  for (int e = t; e < 64 * C0P; e += 256) {
    int n = e / C0P, k = e % C0P;
    Wl[e] = (k < CF) ? (_Float16)W0[n * CF + k] : (_Float16)0.0f;
  }
  __syncthreads();
  const int wave = t >> 5, lane = t & 31;
  const int mbase = (blockIdx.x * 8 + wave) * 16;
  const int mrow = mbase + (lane & 15);
  const int kb   = (lane >> 4) * 8;     // A-frag lane K split
  const int kbB  = (lane >> 4) * 8;     // B-frag lane K split (same as A)
  const int moff = (lane >> 4) * 8;     // C/D row offset
  v16h a[3];
#pragma unroll
  for (int kt = 0; kt < 3; ++kt) {
    const _Float16* p = X0 + (size_t)mrow * C0P + kt * 32 + kb;
    a[kt] = pack_a(*(const v8h*)p, *(const v8h*)(p + 16));
  }
#pragma unroll
  for (int nt = 0; nt < 4; ++nt) {
    const int n = nt * 16 + (lane & 15);
    v8f acc = {};
#pragma unroll
    for (int kt = 0; kt < 3; ++kt) {
      const _Float16* q = &Wl[(size_t)n * C0P + kt * 32 + kbB];
      v16h bf = pack_a(*(const v8h*)q, *(const v8h*)(q + 16));
      acc = __builtin_amdgcn_wmma_f32_16x16x32_f16(false, a[kt], false, bf,
                                                   (short)0, acc, false, false);
    }
    float bb = bias[n];
    float sm = 0.f, sq = 0.f;
#pragma unroll
    for (int r = 0; r < 8; ++r) {
      float v = acc[r] + bb;
      Yout[(size_t)(mbase + moff + r) * 64 + n] = v;
      sm += v; sq += v * v;
    }
    atomicAdd(&sums[n], sm);
    atomicAdd(&sqs[n], sq);
  }
}

// ---------------------------------------------------------------------------
// 4) Layer 1: Y1 = relu(BN0(Y0)) * W1^T + b1 ; BN0 fused into A-frag build.
// ---------------------------------------------------------------------------
__global__ __launch_bounds__(256)
void gemm1_kernel(const float* __restrict__ Yin,   // [M][64]
                  const float* __restrict__ W,     // [64][64]
                  const float* __restrict__ bias,
                  const float* __restrict__ mean_p, const float* __restrict__ rstd_p,
                  const float* __restrict__ g_p,    const float* __restrict__ be_p,
                  float* __restrict__ Yout,         // [M][64]
                  float* __restrict__ sums, float* __restrict__ sqs) {
  __shared__ __align__(32) _Float16 Wl[64 * 64];
  __shared__ float scl[64], sft[64];
  const int t = threadIdx.x;
  for (int e = t; e < 64 * 64; e += 256) Wl[e] = (_Float16)W[e];
  if (t < 64) {
    float sc = rstd_p[t] * g_p[t];
    scl[t] = sc;
    sft[t] = be_p[t] - mean_p[t] * sc;
  }
  __syncthreads();
  const int wave = t >> 5, lane = t & 31;
  const int mbase = (blockIdx.x * 8 + wave) * 16;
  const int mrow = mbase + (lane & 15);
  const int kb = (lane >> 4) * 8, kbB = (lane >> 4) * 16, moff = (lane >> 4) * 8;
  v16h a[2];
#pragma unroll
  for (int kt = 0; kt < 2; ++kt) {
    int k0 = kt * 32 + kb;
    v8f ylo = *(const v8f*)(Yin + (size_t)mrow * 64 + k0);
    v8f yhi = *(const v8f*)(Yin + (size_t)mrow * 64 + k0 + 16);
#pragma unroll
    for (int j = 0; j < 8; ++j) {
      a[kt][j]     = (_Float16)fmaxf(0.f, ylo[j] * scl[k0 + j]      + sft[k0 + j]);
      a[kt][8 + j] = (_Float16)fmaxf(0.f, yhi[j] * scl[k0 + 16 + j] + sft[k0 + 16 + j]);
    }
  }
#pragma unroll
  for (int nt = 0; nt < 4; ++nt) {
    const int n = nt * 16 + (lane & 15);
    v8f acc = {};
#pragma unroll
    for (int kt = 0; kt < 2; ++kt) {
      const _Float16* q = &Wl[(size_t)n * 64 + kt * 32 + kb];
      v16h bf = pack_a(*(const v8h*)q, *(const v8h*)(q + 16));
      acc = __builtin_amdgcn_wmma_f32_16x16x32_f16(false, a[kt], false, bf,
                                                   (short)0, acc, false, false);
    }
    float bb = bias[n];
    float sm = 0.f, sq = 0.f;
#pragma unroll
    for (int r = 0; r < 8; ++r) {
      float v = acc[r] + bb;
      Yout[(size_t)(mbase + moff + r) * 64 + n] = v;
      sm += v; sq += v * v;
    }
    atomicAdd(&sums[n], sm);
    atomicAdd(&sqs[n], sq);
  }
}

// ---------------------------------------------------------------------------
// 5) Layer 2: one wave owns a full (b,s) group (K=32 rows). Computes
//    y2 = relu(BN1(Y1)) * W2^T + b2, BN2 stats, and per-(b,s,c) max & min
//    over K (min kept so the final affine can flip if scale < 0).
// ---------------------------------------------------------------------------
__global__ __launch_bounds__(256)
void gemm2_kernel(const float* __restrict__ Yin,   // [M][64]
                  const float* __restrict__ W,     // [128][64]
                  const float* __restrict__ bias,
                  const float* __restrict__ mean_p, const float* __restrict__ rstd_p,
                  const float* __restrict__ g_p,    const float* __restrict__ be_p,
                  float* __restrict__ sums, float* __restrict__ sqs,
                  float* __restrict__ maxbuf, float* __restrict__ minbuf) {
  __shared__ __align__(32) _Float16 Wl[128 * 64];
  __shared__ float scl[64], sft[64];
  const int t = threadIdx.x;
  for (int e = t; e < 128 * 64; e += 256) Wl[e] = (_Float16)W[e];
  if (t < 64) {
    float sc = rstd_p[t] * g_p[t];
    scl[t] = sc;
    sft[t] = be_p[t] - mean_p[t] * sc;
  }
  __syncthreads();
  const int wave = t >> 5, lane = t & 31;
  const int grp = blockIdx.x * 8 + wave;           // (b*S+s)
  const int kb = (lane >> 4) * 8, kbB = (lane >> 4) * 16;
  v16h a[2][2];
#pragma unroll
  for (int mt = 0; mt < 2; ++mt) {
    int mrow = grp * 32 + mt * 16 + (lane & 15);
#pragma unroll
    for (int kt = 0; kt < 2; ++kt) {
      int k0 = kt * 32 + kb;
      v8f ylo = *(const v8f*)(Yin + (size_t)mrow * 64 + k0);
      v8f yhi = *(const v8f*)(Yin + (size_t)mrow * 64 + k0 + 16);
#pragma unroll
      for (int j = 0; j < 8; ++j) {
        a[mt][kt][j]     = (_Float16)fmaxf(0.f, ylo[j] * scl[k0 + j]      + sft[k0 + j]);
        a[mt][kt][8 + j] = (_Float16)fmaxf(0.f, yhi[j] * scl[k0 + 16 + j] + sft[k0 + 16 + j]);
      }
    }
  }
#pragma unroll
  for (int nt = 0; nt < 8; ++nt) {
    const int n = nt * 16 + (lane & 15);
    v8f accA = {}, accB = {};
#pragma unroll
    for (int kt = 0; kt < 2; ++kt) {
      const _Float16* q = &Wl[(size_t)n * 64 + kt * 32 + kb];
      v16h bf = pack_a(*(const v8h*)q, *(const v8h*)(q + 16));
      accA = __builtin_amdgcn_wmma_f32_16x16x32_f16(false, a[0][kt], false, bf,
                                                    (short)0, accA, false, false);
      accB = __builtin_amdgcn_wmma_f32_16x16x32_f16(false, a[1][kt], false, bf,
                                                    (short)0, accB, false, false);
    }
    float bb = bias[n];
    float sm = 0.f, sq = 0.f, mx = -1e30f, mn = 1e30f;
#pragma unroll
    for (int r = 0; r < 8; ++r) {
      float v0 = accA[r] + bb, v1 = accB[r] + bb;
      sm += v0 + v1; sq += v0 * v0 + v1 * v1;
      mx = fmaxf(mx, fmaxf(v0, v1));
      mn = fminf(mn, fminf(v0, v1));
    }
    atomicAdd(&sums[n], sm);
    atomicAdd(&sqs[n], sq);
    mx = fmaxf(mx, __shfl_xor(mx, 16, 32));
    mn = fminf(mn, __shfl_xor(mn, 16, 32));
    if (lane < 16) {
      maxbuf[(size_t)grp * 128 + n] = mx;
      minbuf[(size_t)grp * 128 + n] = mn;
    }
  }
}

// ---------------------------------------------------------------------------
// 6) BN stats finalize (per layer): mean, rstd from sum/sumsq.
// ---------------------------------------------------------------------------
__global__ void finalize_kernel(const float* __restrict__ sums,
                                const float* __restrict__ sqs,
                                float* __restrict__ mean, float* __restrict__ rstd,
                                int C) {
  int c = threadIdx.x;
  if (c < C) {
    const float inv = 1.0f / (float)M_TOT;
    float mu  = sums[c] * inv;
    float var = sqs[c] * inv - mu * mu;
    mean[c] = mu;
    rstd[c] = rsqrtf(fmaxf(var, 0.f) + BN_EPS);
  }
}

// ---------------------------------------------------------------------------
// 7) Final: out[b,c,s] = relu((sel - mu2)*rstd2*g2 + be2), sel = max or min.
// ---------------------------------------------------------------------------
__global__ __launch_bounds__(256)
void out_kernel(const float* __restrict__ maxbuf, const float* __restrict__ minbuf,
                const float* __restrict__ mean2, const float* __restrict__ rstd2,
                const float* __restrict__ g2, const float* __restrict__ be2,
                float* __restrict__ out) {
  int i = blockIdx.x * 256 + threadIdx.x;          // over B*S*128
  int c = i & 127, p = i >> 7;
  int b = p / SS, s = p % SS;
  float sc = rstd2[c] * g2[c];
  float sel = (sc >= 0.f) ? maxbuf[i] : minbuf[i];
  float v = fmaxf(0.f, (sel - mean2[c]) * sc + be2[c]);
  out[((size_t)b * 128 + c) * SS + s] = v;
}

// ---------------------------------------------------------------------------
extern "C" void kernel_launch(void* const* d_in, const int* in_sizes, int n_in,
                              void* d_out, int out_size, void* d_ws, size_t ws_size,
                              hipStream_t stream) {
  (void)in_sizes; (void)n_in; (void)out_size; (void)ws_size;
  const float* x   = (const float*)d_in[0];
  const float* xc  = (const float*)d_in[1];
  const float* W0  = (const float*)d_in[2];
  const float* b0  = (const float*)d_in[3];
  const float* g0  = (const float*)d_in[4];
  const float* be0 = (const float*)d_in[5];
  const float* W1  = (const float*)d_in[6];
  const float* b1  = (const float*)d_in[7];
  const float* g1  = (const float*)d_in[8];
  const float* be1 = (const float*)d_in[9];
  const float* W2  = (const float*)d_in[10];
  const float* b2  = (const float*)d_in[11];
  const float* g2  = (const float*)d_in[12];
  const float* be2 = (const float*)d_in[13];

  char* ws = (char*)d_ws;
  int*      fpsIdx = (int*)ws;                                  //  64 KB
  float*    stats  = (float*)(ws + 65536);                      //   4 KB
  _Float16* X0     = (_Float16*)(ws + 69632);                   // 96 MiB
  float*    Y0     = (float*)(ws + 69632 + 100663296ull);       // 128 MiB
  float*    Y1     = (float*)(ws + 69632 + 100663296ull + 134217728ull);
  float*    maxbuf = (float*)(ws + 69632 + 100663296ull + 2ull * 134217728ull);
  float*    minbuf = maxbuf + (size_t)BB * SS * 128;

  float* s0 = stats,       *q0 = stats + 64;
  float* s1 = stats + 128, *q1 = stats + 192;
  float* s2 = stats + 256, *q2 = stats + 384;
  float* mean0 = stats + 512, *rstd0 = stats + 576;
  float* mean1 = stats + 640, *rstd1 = stats + 704;
  float* mean2 = stats + 768, *rstd2 = stats + 896;

  (void)hipMemsetAsync(stats, 0, 512 * sizeof(float), stream);

  fps_kernel<<<BB, 1024, 0, stream>>>(x, fpsIdx);
  group_gather_kernel<<<(BB * SS) / 8, 256, 0, stream>>>(x, xc, fpsIdx, X0);
  gemm0_kernel<<<M_TOT / 128, 256, 0, stream>>>(X0, W0, b0, Y0, s0, q0);
  finalize_kernel<<<1, 64, 0, stream>>>(s0, q0, mean0, rstd0, 64);
  gemm1_kernel<<<M_TOT / 128, 256, 0, stream>>>(Y0, W1, b1, mean0, rstd0, g0, be0,
                                                Y1, s1, q1);
  finalize_kernel<<<1, 64, 0, stream>>>(s1, q1, mean1, rstd1, 64);
  gemm2_kernel<<<(BB * SS) / 8, 256, 0, stream>>>(Y1, W2, b2, mean1, rstd1, g1, be1,
                                                  s2, q2, maxbuf, minbuf);
  finalize_kernel<<<1, 128, 0, stream>>>(s2, q2, mean2, rstd2, 128);
  out_kernel<<<(BB * SS * 128) / 256, 256, 0, stream>>>(maxbuf, minbuf, mean2, rstd2,
                                                        g2, be2, (float*)d_out);
}
